// SAGELayer_12077448036513
// MI455X (gfx1250) — matrix operand order repaired and checked by reading, not compiled, not run on hardware
//
#include <hip/hip_runtime.h>

typedef __attribute__((ext_vector_type(16))) _Float16 v16h;
typedef __attribute__((ext_vector_type(8)))  _Float16 v8h;
typedef __attribute__((ext_vector_type(8)))  float    v8f;

#define NN   50000
#define NE   800000
#define DIN  128
#define DHID 128
#define KAPP 256   // 128 nfeat + 128 mean
#define KEDG 256   // 128 h[u] + 128 h[v]
#define EOUT 256

// ---------------- fragment loaders (wave32, v_wmma_f32_16x16x32_f16) -------
// A 16x32 f16: lane L -> row M=L%16; lohi=L/16 selects K halves
//   elements 0..7  = K  kb + 8*lohi      .. +7
//   elements 8..15 = K  kb + 16 + 8*lohi .. +7
__device__ __forceinline__ v16h ld_a_frag(const _Float16* row, int kb, int lohi) {
    v8h lo = *(const v8h*)(row + kb + 8 * lohi);
    v8h hi = *(const v8h*)(row + kb + 16 + 8 * lohi);
    return __builtin_shufflevector(lo, hi, 0,1,2,3,4,5,6,7,8,9,10,11,12,13,14,15);
}
// B 32x16 f16: lane L -> col N=L%16; K = kb + 8*lohi + (0..7), kb + 16 + 8*lohi + (0..7)
__device__ __forceinline__ v16h ld_b_frag(const _Float16* row, int kb, int lohi) {
    v8h lo = *(const v8h*)(row + kb + 8 * lohi);
    v8h hi = *(const v8h*)(row + kb + 16 + 8 * lohi);
    return __builtin_shufflevector(lo, hi, 0,1,2,3,4,5,6,7,8,9,10,11,12,13,14,15);
}

// ---------------- kernel 1: zero segment sums + counts ----------------------
__global__ void zero_kernel(float* p, int n) {
    int i = blockIdx.x * blockDim.x + threadIdx.x;
    int stride = gridDim.x * blockDim.x;
    for (; i < n; i += stride) p[i] = 0.0f;
}

// ---------------- kernel 2: weights f32 -> f16 -------------------------------
__global__ void cvt_kernel(const float* Waw, const float* Wew,
                           _Float16* Wa16, _Float16* We16) {
    int i = blockIdx.x * blockDim.x + threadIdx.x;
    if (i < DHID * KAPP) {
        Wa16[i] = (_Float16)Waw[i];
    } else if (i < DHID * KAPP + EOUT * KEDG) {
        int j = i - DHID * KAPP;
        We16[j] = (_Float16)Wew[j];
    }
}

// ---------------- kernel 3: segment-sum scatter (32 lanes / edge) -----------
__global__ __launch_bounds__(256) void scatter_kernel(const float* __restrict__ efeats,
                                                      const int* __restrict__ v,
                                                      float* __restrict__ sums,
                                                      float* __restrict__ counts) {
    int gid = blockIdx.x * blockDim.x + threadIdx.x;
    int e = gid >> 5;
    int lane = gid & 31;
    if (e >= NE) return;
    int dst = v[e];
    float4 val = *(const float4*)(efeats + (size_t)e * DIN + lane * 4);
    float* s = sums + (size_t)dst * DIN + lane * 4;
    atomicAdd(s + 0, val.x);
    atomicAdd(s + 1, val.y);
    atomicAdd(s + 2, val.z);
    atomicAdd(s + 3, val.w);
    if (lane == 0) atomicAdd(counts + dst, 1.0f);
}

// ---------------- kernel 4: node update GEMM (WMMA f16 -> f32) --------------
// block = 256 thr = 8 waves; 16 nodes per block; each wave owns one 16x16
// output tile (8 waves * 16 = 128 output cols). K = 256, 8 wmma steps.
__global__ __launch_bounds__(256) void node_kernel(const float* __restrict__ nfeats,
                                                   const float* __restrict__ sums,
                                                   const float* __restrict__ counts,
                                                   const _Float16* __restrict__ Wa16,
                                                   const float* __restrict__ Wab,
                                                   float* __restrict__ out_h,
                                                   _Float16* __restrict__ h16) {
    __shared__ _Float16 As[16][264];   // 16 rows x 256 K (+8 pad), 16B-aligned rows

    // ---- stage A = [f16(nfeats) | f16(sums/max(cnt,1))] into LDS (fused mean)
    {
        int t = threadIdx.x;
        int row = t >> 4, c = t & 15;
        int node = blockIdx.x * 16 + row;
        int col0 = c * 16;
        v8h lo, hi;
        if (c < 8) {
            const float* src = nfeats + (size_t)node * DIN + col0;
            #pragma unroll
            for (int i = 0; i < 8; ++i) { lo[i] = (_Float16)src[i]; hi[i] = (_Float16)src[i + 8]; }
        } else {
            float inv = 1.0f / fmaxf(counts[node], 1.0f);
            const float* src = sums + (size_t)node * DIN + (col0 - DIN);
            #pragma unroll
            for (int i = 0; i < 8; ++i) { lo[i] = (_Float16)(src[i] * inv); hi[i] = (_Float16)(src[i + 8] * inv); }
        }
        *(v8h*)&As[row][col0]     = lo;
        *(v8h*)&As[row][col0 + 8] = hi;
    }
    __syncthreads();

    int lane = threadIdx.x & 31;
    int w    = threadIdx.x >> 5;
    int m    = lane & 15;
    int lohi = lane >> 4;
    const _Float16* Arow = &As[m][0];
    const _Float16* Wrow = Wa16 + (size_t)(w * 16 + m) * KAPP;

    v8f acc = {};
    #pragma unroll
    for (int kk = 0; kk < 8; ++kk) {
        int kb = kk * 32;
        v16h a = ld_a_frag(Arow, kb, lohi);
        v16h b = ld_b_frag(Wrow, kb, lohi);
        acc = __builtin_amdgcn_wmma_f32_16x16x32_f16(false, a, false, b, (short)0, acc, false, false);
    }

    int ncol  = w * 16 + m;
    float bias = Wab[ncol];
    int mbase = lohi * 8;
    #pragma unroll
    for (int r = 0; r < 8; ++r) {
        int node = blockIdx.x * 16 + mbase + r;
        float val = fmaxf(acc[r] + bias, 0.0f);
        out_h[(size_t)node * DHID + ncol] = val;
        h16 [(size_t)node * DHID + ncol] = (_Float16)val;
    }
}

// ---------------- kernel 5: edge MLP GEMM (WMMA f16 -> f32) -----------------
// block = 256 thr = 8 waves; 64 edges per block; each wave owns TWO 16x16
// N-tiles (8 waves * 32 = 256 output cols) and FOUR 16-row M-tiles, so every
// weight fragment is reused by 4 WMMAs (4x less W_edge streaming from L2).
__global__ __launch_bounds__(256) void edge_kernel(const _Float16* __restrict__ h16,
                                                   const int* __restrict__ u,
                                                   const int* __restrict__ v,
                                                   const _Float16* __restrict__ We16,
                                                   const float* __restrict__ Web,
                                                   float* __restrict__ out_e) {
    __shared__ _Float16 As[64][264];   // 64 rows x 256 K (+8 pad) = 33 KB

    {   // stage A = [h16[u[e]] | h16[v[e]]]: 1024 32B-chunks, 4 per thread
        int t = threadIdx.x;
        int c = t & 15;
        int col0 = c * 16;
        #pragma unroll
        for (int rr = 0; rr < 4; ++rr) {
            int row = (t >> 4) + rr * 16;
            int e = blockIdx.x * 64 + row;
            const _Float16* src;
            if (col0 < DHID) src = h16 + (size_t)u[e] * DHID + col0;
            else             src = h16 + (size_t)v[e] * DHID + (col0 - DHID);
            *(v8h*)&As[row][col0]     = *(const v8h*)src;
            *(v8h*)&As[row][col0 + 8] = *(const v8h*)(src + 8);
        }
    }
    __syncthreads();

    int lane = threadIdx.x & 31;
    int w    = threadIdx.x >> 5;
    int m    = lane & 15;
    int lohi = lane >> 4;
    const _Float16* Wr0 = We16 + (size_t)(w * 32 + m) * KEDG;
    const _Float16* Wr1 = Wr0 + (size_t)16 * KEDG;

    v8f acc[4][2] = {};
    #pragma unroll
    for (int kk = 0; kk < 8; ++kk) {
        int kb = kk * 32;
        v16h b0 = ld_b_frag(Wr0, kb, lohi);
        v16h b1 = ld_b_frag(Wr1, kb, lohi);
        #pragma unroll
        for (int mt = 0; mt < 4; ++mt) {
            v16h a = ld_a_frag(&As[mt * 16 + m][0], kb, lohi);
            acc[mt][0] = __builtin_amdgcn_wmma_f32_16x16x32_f16(false, a, false, b0, (short)0, acc[mt][0], false, false);
            acc[mt][1] = __builtin_amdgcn_wmma_f32_16x16x32_f16(false, a, false, b1, (short)0, acc[mt][1], false, false);
        }
    }

    int n0 = w * 32 + m;
    float bias0 = Web[n0];
    float bias1 = Web[n0 + 16];
    int mbase = lohi * 8;
    #pragma unroll
    for (int mt = 0; mt < 4; ++mt) {
        #pragma unroll
        for (int r = 0; r < 8; ++r) {
            size_t e = (size_t)blockIdx.x * 64 + mt * 16 + mbase + r;
            out_e[e * EOUT + n0]      = acc[mt][0][r] + bias0;
            out_e[e * EOUT + n0 + 16] = acc[mt][1][r] + bias1;
        }
    }
}

// ---------------- host-side launcher ----------------------------------------
extern "C" void kernel_launch(void* const* d_in, const int* in_sizes, int n_in,
                              void* d_out, int out_size, void* d_ws, size_t ws_size,
                              hipStream_t stream) {
    const float* nfeats = (const float*)d_in[0];
    const float* efeats = (const float*)d_in[1];
    const int*   u      = (const int*)d_in[2];
    const int*   v      = (const int*)d_in[3];
    const float* Waw    = (const float*)d_in[4];
    const float* Wab    = (const float*)d_in[5];
    const float* Wew    = (const float*)d_in[6];
    const float* Web    = (const float*)d_in[7];

    // workspace layout (all 16B aligned)
    char* ws = (char*)d_ws;
    float*     sums   = (float*)ws;                                  // NN*128 f32
    float*     counts = (float*)(ws + (size_t)NN * DIN * 4);         // NN f32 (contiguous after sums)
    _Float16*  h16    = (_Float16*)(ws + (size_t)NN * (DIN + 1) * 4);// NN*128 f16
    _Float16*  Wa16   = (_Float16*)((char*)h16 + (size_t)NN * DHID * 2);
    _Float16*  We16   = Wa16 + DHID * KAPP;

    float* out_h = (float*)d_out;                 // [NN,1,128]
    float* out_e = out_h + (size_t)NN * DHID;     // [NE,1,256]

    // 1) zero sums + counts (contiguous NN*129 floats)
    zero_kernel<<<4096, 256, 0, stream>>>(sums, NN * (DIN + 1));
    // 2) weights -> f16
    int ncvt = DHID * KAPP + EOUT * KEDG;
    cvt_kernel<<<(ncvt + 255) / 256, 256, 0, stream>>>(Waw, Wew, Wa16, We16);
    // 3) segment-sum scatter: 32 lanes per edge
    scatter_kernel<<<(NE * 32) / 256, 256, 0, stream>>>(efeats, v, sums, counts);
    // 4) node update GEMM: 16 nodes / block
    node_kernel<<<NN / 16, 256, 0, stream>>>(nfeats, sums, counts, Wa16, Wab, out_h, h16);
    // 5) edge MLP GEMM: 64 edges / block, 4x weight reuse
    edge_kernel<<<NE / 64, 256, 0, stream>>>(h16, u, v, We16, Web, out_e);
}
